// CPDConvolution2D_82317343195158
// MI455X (gfx1250) — compile-verified, matrix operand order checked by reading, not run on hardware
//
#include <hip/hip_runtime.h>

typedef __attribute__((ext_vector_type(16))) __bf16 v16bf;
typedef __attribute__((ext_vector_type(8)))  __bf16 v8bf;
typedef __attribute__((ext_vector_type(8)))  float  v8f;

#define N_IMG 16
#define S_CH  64
#define R_CH  32
#define T_CH  128
#define H_IMG 224
#define W_IMG 224
#define PLANE (H_IMG * W_IMG)      // 50176
#define TILE_W 16
#define TILE_H 4
#define WT_CNT (W_IMG / TILE_W)    // 14
#define HT_CNT (H_IMG / TILE_H)    // 56
#define WAVES_PER_BLOCK 4
#define SLOT_W 18                  // TILE_W + 2 halo
#define SLOT_H 6                   // TILE_H + 2 halo
#define SLOTS_USED (SLOT_W * SLOT_H)   // 108
#define SLOTS_PAD 112                  // 7 N-tiles of 16
#define NTILES 7

__device__ __forceinline__ v16bf cat8(v8bf lo, v8bf hi) {
    return __builtin_shufflevector(lo, hi, 0,1,2,3,4,5,6,7,8,9,10,11,12,13,14,15);
}

__global__ __launch_bounds__(128) void cpd_fused_kernel(
    const float* __restrict__ x,     // [N,S,H,W]
    const float* __restrict__ w1,    // [R,S]
    const float* __restrict__ dv,    // [R,3]
    const float* __restrict__ dh,    // [R,3]
    const float* __restrict__ w2,    // [T,R]
    const float* __restrict__ bias,  // [T]
    float* __restrict__ out)         // [N,T,H,W]
{
    // ---- LDS: weights (shared) + per-wave staging (total ~58.6 KB) ----
    __shared__ __align__(16) __bf16 w1Lds[R_CH * S_CH];            // 4 KB
    __shared__ __align__(16) __bf16 w2Lds[T_CH * R_CH];            // 8 KB
    __shared__ float dvLds[R_CH * 3];
    __shared__ float dhLds[R_CH * 3];
    __shared__ float biasLds[T_CH];
    __shared__ __align__(16) __bf16 h1Lds[WAVES_PER_BLOCK][R_CH * SLOTS_PAD];     // 4*7 KB
    __shared__ __align__(16) __bf16 h3Lds[WAVES_PER_BLOCK][TILE_H * TILE_W * R_CH]; // 4*4 KB

    const int tid  = threadIdx.x;
    const int lane = tid & 31;
    const int wv   = tid >> 5;
    const int half = lane >> 4;      // 0: lanes 0-15, 1: lanes 16-31
    const int col  = lane & 15;      // N index / M index within 16-tile

    // ---- cooperative weight preload (fp32 -> bf16 for GEMM operands) ----
    for (int i = tid; i < R_CH * S_CH; i += 128) w1Lds[i] = (__bf16)w1[i];
    for (int i = tid; i < T_CH * R_CH; i += 128) w2Lds[i] = (__bf16)w2[i];
    if (tid < R_CH * 3) { dvLds[tid] = dv[tid]; dhLds[tid] = dh[tid]; }
    if (tid < T_CH) biasLds[tid] = bias[tid];
    __syncthreads();

    // ---- tile assignment: one wave = one (n, 4-row, 16-col) output tile ----
    const int gwave = blockIdx.x * WAVES_PER_BLOCK + wv;
    const int wt = gwave % WT_CNT;
    const int ht = (gwave / WT_CNT) % HT_CNT;
    const int n  = gwave / (WT_CNT * HT_CNT);
    const int w0 = wt * TILE_W;
    const int h0 = ht * TILE_H;

    __bf16* h1w = h1Lds[wv];
    __bf16* h3w = h3Lds[wv];

    // ================= Stage 1: h1 = W1(32x64) @ x over halo tile =================
    // N-tiles enumerate 112 pixel-slots covering (6 rows x 18 px) halo region.
    for (int tile = 0; tile < NTILES; ++tile) {
        const int p  = tile * 16 + col;          // slot id for this lane
        const int y  = p / SLOT_W;
        const int px = p - y * SLOT_W;
        const int hh = h0 - 1 + y;
        const int ww = w0 - 1 + px;
        const bool ok = (p < SLOTS_USED) & (hh >= 0) & (hh < H_IMG) & (ww >= 0) & (ww < W_IMG);
        const float msk = ok ? 1.0f : 0.0f;
        const int hc = min(max(hh, 0), H_IMG - 1);
        const int wc = min(max(ww, 0), W_IMG - 1);
        const float* xs = x + n * (S_CH * PLANE) + hc * W_IMG + wc;

        v8f acc0 = {0.f,0.f,0.f,0.f,0.f,0.f,0.f,0.f};
        v8f acc1 = {0.f,0.f,0.f,0.f,0.f,0.f,0.f,0.f};

        #pragma unroll
        for (int ks = 0; ks < 2; ++ks) {          // K = 64 in two 32-chunks
            // B fragment: 16-bit B layout, elem e -> K = (e<8?0:16)+half*8+(e%8)
            v16bf b;
            #pragma unroll
            for (int e = 0; e < 16; ++e) {
                const int K = ((e & 8) << 1) + half * 8 + (e & 7);
                const int s = ks * 32 + K;
                b[e] = (__bf16)(msk * xs[s * PLANE]);
            }
            // A fragments from W1 (rows rb+col), two 8-bf16 runs per lane
            {
                const int base = (0 + col) * S_CH + ks * 32 + half * 8;
                v8bf lo = *(const v8bf*)(w1Lds + base);
                v8bf hi = *(const v8bf*)(w1Lds + base + 16);
                acc0 = __builtin_amdgcn_wmma_f32_16x16x32_bf16(
                           false, cat8(lo, hi), false, b, (short)0, acc0, false, false);
            }
            {
                const int base = (16 + col) * S_CH + ks * 32 + half * 8;
                v8bf lo = *(const v8bf*)(w1Lds + base);
                v8bf hi = *(const v8bf*)(w1Lds + base + 16);
                acc1 = __builtin_amdgcn_wmma_f32_16x16x32_bf16(
                           false, cat8(lo, hi), false, b, (short)0, acc1, false, false);
            }
        }
        // D layout: lane (half,col), VGPR v -> r = rb + v + 8*half, slot = p
        #pragma unroll
        for (int v = 0; v < 8; ++v) {
            h1w[(0  + v + 8 * half) * SLOTS_PAD + p] = (__bf16)acc0[v];
            h1w[(16 + v + 8 * half) * SLOTS_PAD + p] = (__bf16)acc1[v];
        }
    }
    __syncthreads();

    // ======= Stage 2: separable depthwise (3x1 then 1x3), f32 VALU, h1 -> h3 =======
    // 2048 values (32 r x 4 ro x 16 c) per wave; 64 iterations, one value/lane.
    for (int it = 0; it < 64; ++it) {
        const int vi = it * 32 + lane;
        const int r  = vi >> 6;
        const int rem = vi & 63;
        const int ro = rem >> 4;
        const int c  = rem & 15;
        const float v0 = dvLds[r * 3 + 0], v1 = dvLds[r * 3 + 1], v2 = dvLds[r * 3 + 2];
        const float g0 = dhLds[r * 3 + 0], g1 = dhLds[r * 3 + 1], g2 = dhLds[r * 3 + 2];
        const __bf16* hp = h1w + r * SLOTS_PAD;
        float acc = 0.0f;
        #pragma unroll
        for (int j = 0; j < 3; ++j) {
            const float c0 = (float)hp[(ro + 0) * SLOT_W + c + j];
            const float c1 = (float)hp[(ro + 1) * SLOT_W + c + j];
            const float c2 = (float)hp[(ro + 2) * SLOT_W + c + j];
            const float g = (j == 0) ? g0 : ((j == 1) ? g1 : g2);
            acc += g * (v0 * c0 + v1 * c1 + v2 * c2);
        }
        // h3 stored r-contiguous so stage-3 B-frags are two ds_load_b128
        h3w[(ro * 16 + c) * R_CH + r] = (__bf16)acc;
    }
    __syncthreads();

    // ========== Stage 3: out = W2(128x32) @ h3 + bias, K=32 in one WMMA ==========
    for (int ro = 0; ro < TILE_H; ++ro) {
        // B fragment from h3: N = col (pixel), K = r
        const int bbase = (ro * 16 + col) * R_CH + half * 8;
        v8bf blo = *(const v8bf*)(h3w + bbase);
        v8bf bhi = *(const v8bf*)(h3w + bbase + 16);
        const v16bf b = cat8(blo, bhi);
        const int hOut = h0 + ro;

        #pragma unroll
        for (int tb = 0; tb < 8; ++tb) {
            const int abase = (tb * 16 + col) * R_CH + half * 8;
            v8bf alo = *(const v8bf*)(w2Lds + abase);
            v8bf ahi = *(const v8bf*)(w2Lds + abase + 16);
            // init accumulator with bias: lane(half,col), VGPR v -> t = tb*16 + v + 8*half
            v8f c;
            #pragma unroll
            for (int v = 0; v < 8; ++v) c[v] = biasLds[tb * 16 + 8 * half + v];
            c = __builtin_amdgcn_wmma_f32_16x16x32_bf16(
                    false, cat8(alo, ahi), false, b, (short)0, c, false, false);

            float* op = out + ((n * T_CH + tb * 16 + 8 * half) * H_IMG + hOut) * W_IMG + w0 + col;
            #pragma unroll
            for (int v = 0; v < 8; ++v) op[v * PLANE] = c[v];
        }
    }
}

extern "C" void kernel_launch(void* const* d_in, const int* in_sizes, int n_in,
                              void* d_out, int out_size, void* d_ws, size_t ws_size,
                              hipStream_t stream) {
    const float* x    = (const float*)d_in[0];
    const float* w1   = (const float*)d_in[1];
    const float* dv   = (const float*)d_in[2];
    const float* dh   = (const float*)d_in[3];
    const float* w2   = (const float*)d_in[4];
    const float* bias = (const float*)d_in[5];
    float* out = (float*)d_out;

    const int total_waves = N_IMG * HT_CNT * WT_CNT;       // 12544
    const int blocks = total_waves / WAVES_PER_BLOCK;      // 3136
    cpd_fused_kernel<<<blocks, WAVES_PER_BLOCK * 32, 0, stream>>>(
        x, w1, dv, dh, w2, bias, out);
}
